// TemporalGraphEncoder_26070451487090
// MI455X (gfx1250) — compile-verified, matrix operand order checked by reading, not run on hardware
//
#include <hip/hip_runtime.h>
#include <math.h>

#define N_NODES 100000
#define N_EDGES 1600000
#define D_NODE 128
#define D_EDGE 64
#define HID 128
#define HEADS 4
#define NLAYERS 3
#define LN_EPS 1e-5f
#define NEG_SLOPE 0.2f
#define E_TOT (N_EDGES + N_NODES)

typedef __attribute__((ext_vector_type(16))) _Float16 v16h;
typedef __attribute__((ext_vector_type(8)))  _Float16 v8h;
typedef __attribute__((ext_vector_type(8)))  float    v8f;

// ---------------- atomic helpers ----------------
__device__ __forceinline__ void atomicMaxF(float* addr, float v) {
  // standard order-preserving trick; correct for mixed signs, init = -inf
  if (v >= 0.0f) atomicMax((int*)addr, __float_as_int(v));
  else           atomicMin((unsigned int*)addr, __float_as_uint(v));
}
__device__ __forceinline__ void atomicAddF(float* addr, float v) {
  unsafeAtomicAdd(addr, v);   // native global_atomic_add_f32
}

// ---------------- weight packing ----------------
// W0t[n][k] = W0[k][n], f16  (B operand stored column-major = rows of B^T)
__global__ void pack_w0(const float* __restrict__ W0, _Float16* __restrict__ W0t) {
  int i = blockIdx.x * blockDim.x + threadIdx.x;
  if (i >= HID * D_NODE) return;
  int n = i / D_NODE, k = i % D_NODE;
  W0t[i] = (_Float16)W0[k * HID + n];
}
// Wgt[l][n][k] = Wg[l][k][n], f16, n in [0,512)
__global__ void pack_wg(const float* __restrict__ Wg, _Float16* __restrict__ Wgt) {
  int i = blockIdx.x * blockDim.x + threadIdx.x;
  if (i >= NLAYERS * HEADS * HID * HID) return;
  int l = i / (HEADS * HID * HID);
  int r = i % (HEADS * HID * HID);
  int n = r / HID, k = r % HID;
  Wgt[i] = (_Float16)Wg[((size_t)l * HID + k) * (HEADS * HID) + n];
}

// wvec[l][k] = mean_j We[l][k][j]; bmean[l] = mean(be[l]); zero ew_sum
__global__ void prep_edge_weights(const float* __restrict__ We, const float* __restrict__ be,
                                  float* __restrict__ wvec, float* __restrict__ bmean,
                                  float* __restrict__ ew_sum) {
  int l = blockIdx.x, k = threadIdx.x;
  const float* w = We + ((size_t)l * D_EDGE + k) * HID;
  float s = 0.f;
  for (int j = 0; j < HID; ++j) s += w[j];
  wvec[l * D_EDGE + k] = s * (1.0f / HID);
  if (k == 0) {
    float bs = 0.f;
    const float* b = be + l * HID;
    for (int j = 0; j < HID; ++j) bs += b[j];
    bmean[l] = bs * (1.0f / HID);
    ew_sum[l] = 0.f;
  }
}

// ew_all[l][e] = edge_attr[e] . wvec[l] + bmean[l];  ew_sum[l] += ...
__global__ void edge_weight_kernel(const float* __restrict__ edge_attr,
                                   const float* __restrict__ wvec, const float* __restrict__ bmean,
                                   float* __restrict__ ew_all, float* __restrict__ ew_sum) {
  __shared__ float sw[NLAYERS * D_EDGE];
  __shared__ float sb[NLAYERS];
  int tid = threadIdx.x;
  if (tid < NLAYERS * D_EDGE) sw[tid] = wvec[tid];
  if (tid < NLAYERS) sb[tid] = bmean[tid];
  __syncthreads();
  int e = blockIdx.x * blockDim.x + tid;
  float part0 = 0.f, part1 = 0.f, part2 = 0.f;
  if (e < N_EDGES) {
    const float* ea = edge_attr + (size_t)e * D_EDGE;
    float d0 = 0.f, d1 = 0.f, d2 = 0.f;
    for (int k = 0; k < D_EDGE; ++k) {
      float v = ea[k];
      d0 += v * sw[k];
      d1 += v * sw[D_EDGE + k];
      d2 += v * sw[2 * D_EDGE + k];
    }
    d0 += sb[0]; d1 += sb[1]; d2 += sb[2];
    ew_all[e] = d0;
    ew_all[(size_t)N_EDGES + e] = d1;
    ew_all[2 * (size_t)N_EDGES + e] = d2;
    part0 = d0; part1 = d1; part2 = d2;
  }
  for (int off = 16; off; off >>= 1) {
    part0 += __shfl_xor(part0, off, 32);
    part1 += __shfl_xor(part1, off, 32);
    part2 += __shfl_xor(part2, off, 32);
  }
  if ((tid & 31) == 0) {
    atomicAddF(&ew_sum[0], part0);
    atomicAddF(&ew_sum[1], part1);
    atomicAddF(&ew_sum[2], part2);
  }
}

// ---------------- WMMA GEMM: C[M,Ncols] = A[M,128] @ B[128,Ncols] ----------------
// A: f32 row-major (converted to f16 in-register). Bt: f16 [Ncols,128] pre-packed.
// mode 0: outF = relu(C + bias)   (node encoder, f32 out)
// mode 1: outH = (f16)C           (GAT linear, f16 out, L2-resident h)
// Per-lane fragment layout per CDNA5 ISA 7.12.2 (16-bit A 16x32):
//   lane = g*16 + r (g in {0,1}); row/col = tile*16 + r; K halves = [kk+8g .. kk+8g+7] and [kk+16+8g ..]
__global__ void gemm_wmma(const float* __restrict__ A, const _Float16* __restrict__ Bt,
                          const float* __restrict__ bias, float* __restrict__ outF,
                          _Float16* __restrict__ outH, int Ncols, int mode) {
  int wave = threadIdx.x >> 5;
  int lane = threadIdx.x & 31;
  int ntile = blockIdx.y * (blockDim.x >> 5) + wave;
  int mtile = blockIdx.x;
  if (ntile * 16 >= Ncols) return;
  int lrow = lane & 15;
  int koff = (lane >> 4) * 8;
  const float*    arow = A  + (size_t)(mtile * 16 + lrow) * D_NODE;
  const _Float16* brow = Bt + (size_t)(ntile * 16 + lrow) * D_NODE;
  v8f c = {0.f, 0.f, 0.f, 0.f, 0.f, 0.f, 0.f, 0.f};
#pragma unroll
  for (int kk = 0; kk < D_NODE; kk += 32) {
    v8f fa0 = *(const v8f*)(arow + kk + koff);
    v8f fa1 = *(const v8f*)(arow + kk + 16 + koff);
    v8h b0  = *(const v8h*)(brow + kk + koff);
    v8h b1  = *(const v8h*)(brow + kk + 16 + koff);
    v16h a, b;
#pragma unroll
    for (int i = 0; i < 8; ++i) {
      a[i]     = (_Float16)fa0[i];
      a[8 + i] = (_Float16)fa1[i];
      b[i]     = b0[i];
      b[8 + i] = b1[i];
    }
    c = __builtin_amdgcn_wmma_f32_16x16x32_f16(false, a, false, b, (short)0, c, false, false);
  }
  int col   = ntile * 16 + lrow;
  int mbase = mtile * 16 + (lane >> 4) * 8;
  if (mode == 0) {
    float bb = bias[col];
#pragma unroll
    for (int v = 0; v < 8; ++v) {
      float y = c[v] + bb;
      outF[(size_t)(mbase + v) * Ncols + col] = y > 0.f ? y : 0.f;
    }
  } else {
#pragma unroll
    for (int v = 0; v < 8; ++v)
      outH[(size_t)(mbase + v) * Ncols + col] = (_Float16)c[v];
  }
}

// a_s[n,h] = h[n,h,:] . att_src[h,:]   (h is f16, L2-resident)
__global__ void attn_dots(const _Float16* __restrict__ h, const float* __restrict__ att_s,
                          const float* __restrict__ att_d, float* __restrict__ a_s,
                          float* __restrict__ a_d) {
  int idx = blockIdx.x * blockDim.x + threadIdx.x;
  if (idx >= N_NODES * HEADS) return;
  int n = idx >> 2, hd = idx & 3;
  const _Float16* hr = h + ((size_t)n * HEADS + hd) * HID;
  const float* as = att_s + hd * HID;
  const float* ad = att_d + hd * HID;
  float s0 = 0.f, s1 = 0.f;
  for (int c = 0; c < HID; ++c) {
    float v = (float)hr[c];
    s0 += v * as[c];
    s1 += v * ad[c];
  }
  a_s[idx] = s0;
  a_d[idx] = s1;
}

__global__ void layer_init(float* __restrict__ acc, float* __restrict__ m,
                           float* __restrict__ denom) {
  int i = blockIdx.x * blockDim.x + threadIdx.x;
  if (i < N_NODES * HID) acc[i] = 0.f;
  if (i < N_NODES * HEADS) { m[i] = -INFINITY; denom[i] = 0.f; }
}

// pass A: logits = leaky_relu(a_s[src] + a_d[dst] + ew); segment max
__global__ void edge_logits(const int* __restrict__ src, const int* __restrict__ dst,
                            const float* __restrict__ ew_all, const float* __restrict__ ew_sum,
                            const float* __restrict__ a_s, const float* __restrict__ a_d,
                            float* __restrict__ logits, float* __restrict__ m) {
  int e = blockIdx.x * blockDim.x + threadIdx.x;
  if (e >= E_TOT) return;
  int s, d; float ew;
  if (e < N_EDGES) { s = src[e]; d = dst[e]; ew = ew_all[e]; }
  else             { s = d = e - N_EDGES; ew = ew_sum[0] * (1.0f / N_EDGES); }
  float4 as = *(const float4*)(a_s + (size_t)s * 4);
  float4 ad = *(const float4*)(a_d + (size_t)d * 4);
  float lg[4] = {as.x + ad.x + ew, as.y + ad.y + ew, as.z + ad.z + ew, as.w + ad.w + ew};
#pragma unroll
  for (int hd = 0; hd < HEADS; ++hd) {
    float v = lg[hd];
    v = v > 0.f ? v : NEG_SLOPE * v;
    logits[(size_t)e * 4 + hd] = v;
    atomicMaxF(&m[(size_t)d * 4 + hd], v);
  }
}

// pass B: ex = exp(logit - m[dst]); segment sum
__global__ void edge_exp(const int* __restrict__ dst, const float* __restrict__ m,
                         float* __restrict__ logits, float* __restrict__ denom) {
  int e = blockIdx.x * blockDim.x + threadIdx.x;
  if (e >= E_TOT) return;
  int d = (e < N_EDGES) ? dst[e] : (e - N_EDGES);
  float4 mm = *(const float4*)(m + (size_t)d * 4);
  float4 lg = *(float4*)(logits + (size_t)e * 4);
  float e0 = __expf(lg.x - mm.x), e1 = __expf(lg.y - mm.y);
  float e2 = __expf(lg.z - mm.z), e3 = __expf(lg.w - mm.w);
  *(float4*)(logits + (size_t)e * 4) = make_float4(e0, e1, e2, e3);
  atomicAddF(&denom[(size_t)d * 4 + 0], e0);
  atomicAddF(&denom[(size_t)d * 4 + 1], e1);
  atomicAddF(&denom[(size_t)d * 4 + 2], e2);
  atomicAddF(&denom[(size_t)d * 4 + 3], e3);
}

// pass C: wave per edge; acc[dst] += sum_h alpha_h * h[src,h,:]  (head sum fused)
__global__ void edge_message(const int* __restrict__ src, const int* __restrict__ dst,
                             const float* __restrict__ ex, const float* __restrict__ denom,
                             const _Float16* __restrict__ h, float* __restrict__ acc) {
  int lane = threadIdx.x & 31;
  int e = (blockIdx.x * blockDim.x + threadIdx.x) >> 5;
  if (e >= E_TOT) return;
  int s, d;
  if (e < N_EDGES) { s = src[e]; d = dst[e]; } else { s = d = e - N_EDGES; }
  float4 exv = *(const float4*)(ex + (size_t)e * 4);
  float4 dn  = *(const float4*)(denom + (size_t)d * 4);
  float alpha[4] = {exv.x / dn.x, exv.y / dn.y, exv.z / dn.z, exv.w / dn.w};
  int c0 = lane * 4;
  float msg[4] = {0.f, 0.f, 0.f, 0.f};
  const _Float16* hs = h + (size_t)s * (HEADS * HID);
#pragma unroll
  for (int hd = 0; hd < HEADS; ++hd) {
    const _Float16* p = hs + hd * HID + c0;   // 8B contiguous
    float a = alpha[hd];
    msg[0] += a * (float)p[0];
    msg[1] += a * (float)p[1];
    msg[2] += a * (float)p[2];
    msg[3] += a * (float)p[3];
  }
  float* ap = acc + (size_t)d * HID + c0;
#pragma unroll
  for (int i = 0; i < 4; ++i) atomicAddF(&ap[i], msg[i]);
}

// wave per node: y = LN(acc/4 + bg); x = relu(x + y)
__global__ void node_finalize(const float* __restrict__ acc, const float* __restrict__ bg,
                              const float* __restrict__ gamma, const float* __restrict__ beta,
                              float* __restrict__ x) {
  int lane = threadIdx.x & 31;
  int n = (blockIdx.x * blockDim.x + threadIdx.x) >> 5;
  if (n >= N_NODES) return;
  float v[4];
  float sum = 0.f;
#pragma unroll
  for (int i = 0; i < 4; ++i) {
    int c = lane + 32 * i;
    float y = acc[(size_t)n * HID + c] * (1.0f / HEADS) + bg[c];
    v[i] = y;
    sum += y;
  }
  for (int off = 16; off; off >>= 1) sum += __shfl_xor(sum, off, 32);
  float mu = sum * (1.0f / HID);
  float var = 0.f;
#pragma unroll
  for (int i = 0; i < 4; ++i) { float t = v[i] - mu; var += t * t; }
  for (int off = 16; off; off >>= 1) var += __shfl_xor(var, off, 32);
  float inv = rsqrtf(var * (1.0f / HID) + LN_EPS);
#pragma unroll
  for (int i = 0; i < 4; ++i) {
    int c = lane + 32 * i;
    float y = (v[i] - mu) * inv * gamma[c] + beta[c];
    float r = x[(size_t)n * HID + c] + y;
    x[(size_t)n * HID + c] = r > 0.f ? r : 0.f;
  }
}

// ---------------- launch ----------------
extern "C" void kernel_launch(void* const* d_in, const int* in_sizes, int n_in,
                              void* d_out, int out_size, void* d_ws, size_t ws_size,
                              hipStream_t stream) {
  (void)in_sizes; (void)n_in; (void)out_size; (void)ws_size;
  const float* NF      = (const float*)d_in[0];
  const int*   EI      = (const int*)  d_in[1];
  const float* EA      = (const float*)d_in[2];
  const float* W0      = (const float*)d_in[3];
  const float* b0      = (const float*)d_in[4];
  const float* We      = (const float*)d_in[5];
  const float* be      = (const float*)d_in[6];
  const float* Wg      = (const float*)d_in[7];
  const float* att_src = (const float*)d_in[8];
  const float* att_dst = (const float*)d_in[9];
  const float* bg      = (const float*)d_in[10];
  const float* gamma   = (const float*)d_in[11];
  const float* beta    = (const float*)d_in[12];
  float* x = (float*)d_out;                 // running node features, [N,128] f32
  const int* src = EI;
  const int* dst = EI + N_EDGES;

  char* ws = (char*)d_ws;
  size_t off = 0;
  auto alloc = [&](size_t bytes) -> char* {
    char* p = ws + off;
    off = (off + bytes + 255) & ~(size_t)255;
    return p;
  };
  _Float16* W0t   = (_Float16*)alloc((size_t)HID * D_NODE * 2);
  _Float16* Wgt   = (_Float16*)alloc((size_t)NLAYERS * HEADS * HID * HID * 2);
  float*    wvec  = (float*)alloc((size_t)NLAYERS * D_EDGE * 4);
  float*    bmean = (float*)alloc((size_t)NLAYERS * 4);
  float*    ewsum = (float*)alloc((size_t)NLAYERS * 4);
  float*    ewall = (float*)alloc((size_t)NLAYERS * N_EDGES * 4);
  _Float16* h     = (_Float16*)alloc((size_t)N_NODES * HEADS * HID * 2);  // 102 MB, L2-resident
  float*    a_s   = (float*)alloc((size_t)N_NODES * HEADS * 4);
  float*    a_d   = (float*)alloc((size_t)N_NODES * HEADS * 4);
  float*    mbuf  = (float*)alloc((size_t)N_NODES * HEADS * 4);
  float*    denom = (float*)alloc((size_t)N_NODES * HEADS * 4);
  float*    logit = (float*)alloc((size_t)E_TOT * HEADS * 4);
  float*    acc   = (float*)alloc((size_t)N_NODES * HID * 4);

  const int B = 256;
  // weight packing + edge-weight precompute (all layers fused)
  pack_w0<<<(HID * D_NODE + B - 1) / B, B, 0, stream>>>(W0, W0t);
  pack_wg<<<(NLAYERS * HEADS * HID * HID + B - 1) / B, B, 0, stream>>>(Wg, Wgt);
  prep_edge_weights<<<NLAYERS, D_EDGE, 0, stream>>>(We, be, wvec, bmean, ewsum);
  edge_weight_kernel<<<(N_EDGES + B - 1) / B, B, 0, stream>>>(EA, wvec, bmean, ewall, ewsum);

  // node encoder: x = relu(NF @ W0 + b0)
  {
    dim3 grid(N_NODES / 16, (HID / 16 + 7) / 8);
    gemm_wmma<<<grid, B, 0, stream>>>(NF, W0t, b0, x, (_Float16*)nullptr, HID, 0);
  }

  for (int l = 0; l < NLAYERS; ++l) {
    // h = (f16)(x @ Wg[l])   [N, 4*128]
    dim3 grid(N_NODES / 16, (HEADS * HID / 16) / 8);
    gemm_wmma<<<grid, B, 0, stream>>>(x, Wgt + (size_t)l * HEADS * HID * HID,
                                      (const float*)nullptr, (float*)nullptr, h,
                                      HEADS * HID, 1);
    attn_dots<<<(N_NODES * HEADS + B - 1) / B, B, 0, stream>>>(
        h, att_src + (size_t)l * HEADS * HID, att_dst + (size_t)l * HEADS * HID, a_s, a_d);
    layer_init<<<(N_NODES * HID + B - 1) / B, B, 0, stream>>>(acc, mbuf, denom);
    edge_logits<<<(E_TOT + B - 1) / B, B, 0, stream>>>(
        src, dst, ewall + (size_t)l * N_EDGES, ewsum + l, a_s, a_d, logit, mbuf);
    edge_exp<<<(E_TOT + B - 1) / B, B, 0, stream>>>(dst, mbuf, logit, denom);
    edge_message<<<((size_t)E_TOT * 32 + B - 1) / B, B, 0, stream>>>(
        src, dst, logit, denom, h, acc);
    node_finalize<<<((size_t)N_NODES * 32 + B - 1) / B, B, 0, stream>>>(
        acc, bg + (size_t)l * HID, gamma + (size_t)l * HID, beta + (size_t)l * HID, x);
  }
}
